// RelativeAttentionCell_15247133901355
// MI455X (gfx1250) — compile-verified, matrix operand order checked by reading, not run on hardware
//
#include <hip/hip_runtime.h>
#include <hip/hip_bf16.h>

// ---------------------------------------------------------------------------
// RelativeAttentionCell on gfx1250 (MI455X), WMMA f16->f32 pipeline.
// B=2 T=512 M=512 KV=1024 D=512 H=8 U=64 O=512, rel table R=1024.
// ---------------------------------------------------------------------------

typedef __attribute__((ext_vector_type(16))) _Float16 v16h;
typedef __attribute__((ext_vector_type(8)))  _Float16 v8h;
typedef __attribute__((ext_vector_type(8)))  float    v8f;

#define BB   2
#define TT   512
#define MM   512
#define DD   512
#define HH   8
#define UU   64
#define OO   512
#define KVV  1024
#define RR   1024

// load a 16x32 f16 fragment whose per-lane base pointer is already resolved
__device__ __forceinline__ v16h ldfrag(const _Float16* __restrict__ q) {
    v8h lo = *(const v8h*)(q);
    v8h hi = *(const v8h*)(q + 16);
    v16h f;
#pragma unroll
    for (int i = 0; i < 8; ++i) { f[i] = lo[i]; f[8 + i] = hi[i]; }
    return f;
}

#define WMMA(a, b, c) __builtin_amdgcn_wmma_f32_16x16x32_f16(false, a, false, b, (short)0, c, false, false)

// ---------------- generic WMMA GEMM ----------------------------------------
// C[m][n] = sum_k A[m][k] * Bt[n][k]   (Bt is B transposed, row-major)
// grid: x = N/64, y = Mrows/256, z = nB*nH ; block = 256 (8 waves).
// Wave tile 32x64 (2 M x 4 N subtiles). K must be a multiple of 64.
// Manual 2-stage ping-pong: fragment sets x*/y* live simultaneously so the
// allocator cannot coalesce them -> loads never clobber in-use WMMA sources.
// band >= 0: skip whole block if n0 >= blockRow0 + 256 + band (causal band).
__global__ void k_gemm_f16(const _Float16* __restrict__ A, long sAb, long sAh, int lda,
                           const _Float16* __restrict__ Bt, long sBb, long sBh, int ldbt,
                           _Float16* __restrict__ Ch, float* __restrict__ Cf,
                           long sCb, long sCh, int ldc, int K, int nH, int band) {
    int n0 = blockIdx.x * 64;
    if (band >= 0 && n0 >= (int)blockIdx.y * 256 + 256 + band) return;

    int z = blockIdx.z;
    int b = z / nH;
    int h = z % nH;
    A  += (long)b * sAb + (long)h * sAh;
    Bt += (long)b * sBb + (long)h * sBh;
    long coff = (long)b * sCb + (long)h * sCh;

    int wave = threadIdx.x >> 5;                 // 0..7
    int m0   = blockIdx.y * 256 + wave * 32;

    // per-lane fragment base pointers (row = base + lane%16, k = (lane/16)*8)
    int lane = threadIdx.x & 31;
    int rr   = lane & 15;
    int kb   = (lane >> 4) << 3;
    const _Float16* la0 = A  + (long)(m0 +  0 + rr) * lda  + kb;
    const _Float16* la1 = A  + (long)(m0 + 16 + rr) * lda  + kb;
    const _Float16* lb0 = Bt + (long)(n0 +  0 + rr) * ldbt + kb;
    const _Float16* lb1 = Bt + (long)(n0 + 16 + rr) * ldbt + kb;
    const _Float16* lb2 = Bt + (long)(n0 + 32 + rr) * ldbt + kb;
    const _Float16* lb3 = Bt + (long)(n0 + 48 + rr) * ldbt + kb;

    v8f c00 = {}, c01 = {}, c02 = {}, c03 = {};
    v8f c10 = {}, c11 = {}, c12 = {}, c13 = {};

    // stage-0 fragments (k = 0)
    v16h xa0 = ldfrag(la0), xa1 = ldfrag(la1);
    v16h xb0 = ldfrag(lb0), xb1 = ldfrag(lb1), xb2 = ldfrag(lb2), xb3 = ldfrag(lb3);
    la0 += 32; la1 += 32; lb0 += 32; lb1 += 32; lb2 += 32; lb3 += 32;

#pragma unroll 1
    for (int k0 = 0; k0 < K - 64; k0 += 64) {
        // stage-1 loads (k0 + 32)
        v16h ya0 = ldfrag(la0), ya1 = ldfrag(la1);
        v16h yb0 = ldfrag(lb0), yb1 = ldfrag(lb1), yb2 = ldfrag(lb2), yb3 = ldfrag(lb3);
        la0 += 32; la1 += 32; lb0 += 32; lb1 += 32; lb2 += 32; lb3 += 32;

        c00 = WMMA(xa0, xb0, c00);  c10 = WMMA(xa1, xb0, c10);
        c01 = WMMA(xa0, xb1, c01);  c11 = WMMA(xa1, xb1, c11);
        c02 = WMMA(xa0, xb2, c02);  c12 = WMMA(xa1, xb2, c12);
        c03 = WMMA(xa0, xb3, c03);  c13 = WMMA(xa1, xb3, c13);

        // stage-0 loads (k0 + 64)
        xa0 = ldfrag(la0); xa1 = ldfrag(la1);
        xb0 = ldfrag(lb0); xb1 = ldfrag(lb1); xb2 = ldfrag(lb2); xb3 = ldfrag(lb3);
        la0 += 32; la1 += 32; lb0 += 32; lb1 += 32; lb2 += 32; lb3 += 32;

        c00 = WMMA(ya0, yb0, c00);  c10 = WMMA(ya1, yb0, c10);
        c01 = WMMA(ya0, yb1, c01);  c11 = WMMA(ya1, yb1, c11);
        c02 = WMMA(ya0, yb2, c02);  c12 = WMMA(ya1, yb2, c12);
        c03 = WMMA(ya0, yb3, c03);  c13 = WMMA(ya1, yb3, c13);
    }

    // tail: last two K-steps (K-64 .. K)
    {
        v16h ya0 = ldfrag(la0), ya1 = ldfrag(la1);
        v16h yb0 = ldfrag(lb0), yb1 = ldfrag(lb1), yb2 = ldfrag(lb2), yb3 = ldfrag(lb3);

        c00 = WMMA(xa0, xb0, c00);  c10 = WMMA(xa1, xb0, c10);
        c01 = WMMA(xa0, xb1, c01);  c11 = WMMA(xa1, xb1, c11);
        c02 = WMMA(xa0, xb2, c02);  c12 = WMMA(xa1, xb2, c12);
        c03 = WMMA(xa0, xb3, c03);  c13 = WMMA(xa1, xb3, c13);

        c00 = WMMA(ya0, yb0, c00);  c10 = WMMA(ya1, yb0, c10);
        c01 = WMMA(ya0, yb1, c01);  c11 = WMMA(ya1, yb1, c11);
        c02 = WMMA(ya0, yb2, c02);  c12 = WMMA(ya1, yb2, c12);
        c03 = WMMA(ya0, yb3, c03);  c13 = WMMA(ya1, yb3, c13);
    }

    int n  = lane & 15;
    int mh = (lane >> 4) * 8;
    v8f accs[2][4] = {{c00, c01, c02, c03}, {c10, c11, c12, c13}};
#pragma unroll
    for (int s = 0; s < 2; ++s) {
#pragma unroll
        for (int j = 0; j < 4; ++j) {
#pragma unroll
            for (int i = 0; i < 8; ++i) {
                long idx = coff + (long)(m0 + s * 16 + mh + i) * ldc + (n0 + j * 16 + n);
                if (Cf) Cf[idx] = accs[s][j][i];
                else    Ch[idx] = (_Float16)accs[s][j][i];
            }
        }
    }
}

// ---------------- elementwise / prep kernels -------------------------------
__global__ void k_transpose_c2h(const float* __restrict__ in, _Float16* __restrict__ out,
                                int rows, int cols) {
    int idx = blockIdx.x * 256 + threadIdx.x;
    if (idx >= rows * cols) return;
    int r = idx / cols, c = idx % cols;
    out[(long)c * rows + r] = (_Float16)in[idx];
}

__global__ void k_copy_c2h(const float* __restrict__ in, _Float16* __restrict__ out, int n) {
    int idx = blockIdx.x * 256 + threadIdx.x;
    if (idx < n) out[idx] = (_Float16)in[idx];
}

__global__ void k_build_xcat(const float* __restrict__ inputs, const float* __restrict__ state,
                             _Float16* __restrict__ out) {
    long idx = (long)blockIdx.x * 256 + threadIdx.x;          // B*KV*D
    if (idx >= (long)BB * KVV * DD) return;
    int  d  = (int)(idx % DD);
    long r  = idx / DD;
    int  kv = (int)(r % KVV);
    int  b  = (int)(r / KVV);
    float v = (kv < MM) ? state[((long)b * MM + kv) * DD + d]
                        : inputs[((long)b * TT + (kv - MM)) * DD + d];
    out[idx] = (_Float16)v;
}

__global__ void k_sintab(_Float16* __restrict__ S) {
    int idx = blockIdx.x * 256 + threadIdx.x;                 // R*D
    if (idx >= RR * DD) return;
    int d = idx % DD, r = idx / DD;
    float w   = __powf(10000.f, (2.f * (float)d) / (float)DD);
    float off = (d & 1) ? 1.57079632679f : 0.f;
    S[idx] = (_Float16)__sinf((float)r / w + off);
}

// mask query, build query+v
__global__ void k_q_post(_Float16* __restrict__ Q, _Float16* __restrict__ Qpv,
                         const unsigned char* __restrict__ mask,
                         const float* __restrict__ vbias) {
    int idx = blockIdx.x * 256 + threadIdx.x;                 // B*T*H*U
    if (idx >= BB * TT * HH * UU) return;
    int j = idx % (HH * UU);
    int t = (idx / (HH * UU)) % TT;
    int b = idx / (TT * HH * UU);
    float mf = (float)mask[b * TT + t];
    float qm = (float)Q[idx] * mf;
    Q[idx]   = (_Float16)qm;
    Qpv[idx] = (_Float16)(qm + vbias[j]);
}

// ubias[b][h][t] = sum_u K[b][t][h][u] * u[h][u]
__global__ void k_ubias(const _Float16* __restrict__ Kp, const float* __restrict__ uw,
                        float* __restrict__ ub) {
    int idx = blockIdx.x * 256 + threadIdx.x;                 // B*H*KV
    if (idx >= BB * HH * KVV) return;
    int t = idx % KVV;
    int h = (idx / KVV) % HH;
    int b = idx / (HH * KVV);
    const _Float16* kp = Kp + ((long)b * KVV + t) * (HH * UU) + h * UU;
    const float*    up = uw + h * UU;
    float s = 0.f;
#pragma unroll 8
    for (int u = 0; u < UU; ++u) s += (float)kp[u] * up[u];
    ub[idx] = s;
}

// Vt[b][h][u][t] = V[b][t][h][u]
__global__ void k_vt(const _Float16* __restrict__ V, _Float16* __restrict__ Vt) {
    int idx = blockIdx.x * 256 + threadIdx.x;                 // B*H*U*KV
    if (idx >= BB * HH * UU * KVV) return;
    int t = idx % KVV;
    int u = (idx / KVV) % UU;
    int h = (idx / (KVV * UU)) % HH;
    int b = idx / (KVV * UU * HH);
    Vt[idx] = V[((long)b * KVV + t) * (HH * UU) + h * UU + u];
}

// ---------------- fused bias/mask/softmax row kernel -----------------------
// one block (256 threads) per (b,h,q) row of KV=1024 logits
__global__ void k_softmax(const float* __restrict__ cc, const float* __restrict__ P,
                          const float* __restrict__ ub,
                          const unsigned char* __restrict__ smask,
                          const unsigned char* __restrict__ qmask,
                          _Float16* __restrict__ W) {
    int q = blockIdx.x, h = blockIdx.y, b = blockIdx.z;
    long row = (((long)b * HH + h) * TT + q) * KVV;
    const float* pc = cc + row;
    const float* pp = P + row;                                 // same (b,h,T,1024) shape
    const float* pu = ub + ((long)b * HH + h) * KVV;

    float l[4];
    float mx = -3.4e38f;
#pragma unroll
    for (int i = 0; i < 4; ++i) {
        int t = threadIdx.x + i * 256;
        int r = q - t + MM;
        r = r < 0 ? 0 : (r > RR - 1 ? RR - 1 : r);
        float lg = (pc[t] + pp[r] + pu[t]) * 0.125f;           // 1/sqrt(64)
        if (t - q >= KVV - TT + 1) lg = -10000.f;              // causal (dominates)
        float cm = (t < MM) ? (float)smask[b * MM + t]
                            : (float)qmask[b * TT + (t - MM)];
        lg -= (1.f - cm) * 10000.f;                            // key padding
        l[i] = lg;
        mx = fmaxf(mx, lg);
    }
    __shared__ float red[8];
#pragma unroll
    for (int o = 16; o > 0; o >>= 1) mx = fmaxf(mx, __shfl_xor(mx, o, 32));
    if ((threadIdx.x & 31) == 0) red[threadIdx.x >> 5] = mx;
    __syncthreads();
    float m2 = red[0];
#pragma unroll
    for (int i = 1; i < 8; ++i) m2 = fmaxf(m2, red[i]);
    __syncthreads();

    float e[4];
    float s = 0.f;
#pragma unroll
    for (int i = 0; i < 4; ++i) { e[i] = __expf(l[i] - m2); s += e[i]; }
#pragma unroll
    for (int o = 16; o > 0; o >>= 1) s += __shfl_xor(s, o, 32);
    if ((threadIdx.x & 31) == 0) red[threadIdx.x >> 5] = s;
    __syncthreads();
    float s2 = 0.f;
#pragma unroll
    for (int i = 0; i < 8; ++i) s2 += red[i];
    float inv = 1.f / s2;
#pragma unroll
    for (int i = 0; i < 4; ++i) {
        int t = threadIdx.x + i * 256;
        W[row + t] = (_Float16)(e[i] * inv);
    }
}

// ---------------------------------------------------------------------------
extern "C" void kernel_launch(void* const* d_in, const int* in_sizes, int n_in,
                              void* d_out, int out_size, void* d_ws, size_t ws_size,
                              hipStream_t stream) {
    const float* inputs   = (const float*)d_in[0];
    const float* state    = (const float*)d_in[1];
    const unsigned char* mask  = (const unsigned char*)d_in[2];
    const unsigned char* smask = (const unsigned char*)d_in[3];
    const float* query_W  = (const float*)d_in[4];
    const float* key_W    = (const float*)d_in[5];
    const float* rel_W    = (const float*)d_in[6];
    const float* value_W  = (const float*)d_in[7];
    const float* u_w      = (const float*)d_in[8];
    const float* v_w      = (const float*)d_in[9];
    const float* output_W = (const float*)d_in[10];
    float* out = (float*)d_out;
    (void)in_sizes; (void)n_in; (void)out_size; (void)ws_size;

    // workspace carve-up (256B aligned)
    char* p = (char*)d_ws;
    auto alloc = [&](size_t bytes) { char* r = p; p += (bytes + 255) & ~(size_t)255; return r; };
    _Float16* Xcat = (_Float16*)alloc((size_t)BB * KVV * DD * 2);
    _Float16* WqT  = (_Float16*)alloc((size_t)DD * HH * UU * 2);
    _Float16* WkT  = (_Float16*)alloc((size_t)DD * HH * UU * 2);
    _Float16* WvT  = (_Float16*)alloc((size_t)DD * HH * UU * 2);
    _Float16* WoT  = (_Float16*)alloc((size_t)HH * UU * OO * 2);
    _Float16* Wr16 = (_Float16*)alloc((size_t)DD * HH * UU * 2);
    _Float16* Stab = (_Float16*)alloc((size_t)RR * DD * 2);
    _Float16* Qh   = (_Float16*)alloc((size_t)BB * TT * HH * UU * 2);
    _Float16* Qpv  = (_Float16*)alloc((size_t)BB * TT * HH * UU * 2);
    _Float16* Kh   = (_Float16*)alloc((size_t)BB * KVV * HH * UU * 2);
    _Float16* Vh   = (_Float16*)alloc((size_t)BB * KVV * HH * UU * 2);
    _Float16* Vt   = (_Float16*)alloc((size_t)BB * HH * UU * KVV * 2);
    _Float16* qv   = (_Float16*)alloc((size_t)BB * HH * TT * DD * 2);
    float*    cc   = (float*)alloc((size_t)BB * HH * TT * KVV * 4);
    float*    Pb   = (float*)alloc((size_t)BB * HH * TT * KVV * 4);
    float*    ub   = (float*)alloc((size_t)BB * HH * KVV * 4);
    _Float16* Wsm  = (_Float16*)alloc((size_t)BB * HH * TT * KVV * 2);
    _Float16* att  = (_Float16*)alloc((size_t)BB * TT * HH * UU * 2);

    // ---- weight conversion (f32 -> f16, B-transposed layouts) ----
    {
        int n = DD * HH * UU;
        int g = (n + 255) / 256;
        k_transpose_c2h<<<g, 256, 0, stream>>>(query_W, WqT, DD, HH * UU);
        k_transpose_c2h<<<g, 256, 0, stream>>>(key_W,   WkT, DD, HH * UU);
        k_transpose_c2h<<<g, 256, 0, stream>>>(value_W, WvT, DD, HH * UU);
        k_transpose_c2h<<<g, 256, 0, stream>>>(output_W, WoT, HH * UU, OO);
        k_copy_c2h<<<g, 256, 0, stream>>>(rel_W, Wr16, n);
    }
    k_build_xcat<<<(BB * KVV * DD + 255) / 256, 256, 0, stream>>>(inputs, state, Xcat);
    k_sintab<<<(RR * DD + 255) / 256, 256, 0, stream>>>(Stab);

    auto gemm = [&](const _Float16* A, long sAb, long sAh, int lda,
                    const _Float16* Bt, long sBb, long sBh, int ldbt,
                    _Float16* Ch, float* Cf, long sCb, long sCh, int ldc,
                    int Mrows, int N, int K, int nB, int nH, int band) {
        dim3 grid(N / 64, Mrows / 256, nB * nH);
        k_gemm_f16<<<grid, 256, 0, stream>>>(A, sAb, sAh, lda, Bt, sBb, sBh, ldbt,
                                             Ch, Cf, sCb, sCh, ldc, K, nH, band);
    };

    const int NH = HH * UU;  // 512
    // Q = inputs-part-of-Xcat @ Wq
    gemm(Xcat + (long)MM * DD, (long)KVV * DD, 0, DD,
         WqT, 0, 0, DD,
         Qh, nullptr, (long)TT * NH, 0, NH,
         TT, NH, DD, BB, 1, -1);
    // K = Xcat @ Wk ; V = Xcat @ Wv
    gemm(Xcat, (long)KVV * DD, 0, DD, WkT, 0, 0, DD,
         Kh, nullptr, (long)KVV * NH, 0, NH, KVV, NH, DD, BB, 1, -1);
    gemm(Xcat, (long)KVV * DD, 0, DD, WvT, 0, 0, DD,
         Vh, nullptr, (long)KVV * NH, 0, NH, KVV, NH, DD, BB, 1, -1);

    // query mask + (query + v)
    k_q_post<<<(BB * TT * NH + 255) / 256, 256, 0, stream>>>(Qh, Qpv, mask, v_w);
    // global content bias term
    k_ubias<<<(BB * HH * KVV + 255) / 256, 256, 0, stream>>>(Kh, u_w, ub);

    // qv[b,h] = (Q+v)[b,:,h,:] @ rel_W[:,h,:]^T   (T x U) x (U x D)
    gemm(Qpv, (long)TT * NH, UU, NH,
         Wr16, 0, UU, NH,                  // Bt row d: rel_W[(d*H+h)*U + u]
         qv, nullptr, (long)HH * TT * DD, (long)TT * DD, DD,
         TT, DD, UU, BB, HH, -1);

    // content-content logits: cc[b,h] = Q[b,:,h,:] @ K[b,:,h,:]^T
    // (tiles fully beyond the causal band are skipped)
    gemm(Qh, (long)TT * NH, UU, NH,
         Kh, (long)KVV * NH, UU, NH,
         nullptr, cc, (long)HH * TT * KVV, (long)TT * KVV, KVV,
         TT, KVV, UU, BB, HH, 512);

    // position logits table: P[b,h] = qv[b,h] @ S^T   (T x D) x (D x R)
    // (r = q - t + 512 <= q + 512, so r-tiles beyond the band are never read)
    gemm(qv, (long)HH * TT * DD, (long)TT * DD, DD,
         Stab, 0, 0, DD,
         nullptr, Pb, (long)HH * TT * RR, (long)TT * RR, RR,
         TT, RR, DD, BB, HH, 512);

    // softmax with causal + padding masks + rel-gather + u-bias
    {
        dim3 grid(TT, HH, BB);
        k_softmax<<<grid, 256, 0, stream>>>(cc, Pb, ub, smask, mask, Wsm);
    }

    // V transpose per head, then attended = softmax @ V
    k_vt<<<(BB * HH * UU * KVV + 255) / 256, 256, 0, stream>>>(Vh, Vt);
    gemm(Wsm, (long)HH * TT * KVV, (long)TT * KVV, KVV,
         Vt, (long)HH * UU * KVV, (long)UU * KVV, KVV,
         att, nullptr, (long)TT * NH, UU, NH,
         TT, UU, KVV, BB, HH, -1);

    // output = attended @ output_W   (B*T x 512) x (512 x 512)
    gemm(att, 0, 0, NH,
         WoT, 0, 0, NH,
         nullptr, out, 0, 0, OO,
         BB * TT, OO, NH, 1, 1, -1);
}
